// SurvivalCopula_4363686773585
// MI455X (gfx1250) — matrix operand order repaired and checked by reading, not compiled, not run on hardware
//
#include <hip/hip_runtime.h>
#include <hip/hip_bf16.h>

// ---------------------------------------------------------------------------
// SurvivalCopula on MI455X (gfx1250, wave32).
//
//  * Linear MLPs folded to per-row 32-dot products (128 B/row read).
//  * Dot products on the matrix pipe: v_wmma_f32_16x16x32_f16 (f32 accum).
//  * Copula math fully closed-form:
//      e      = exp(k) = -logS
//      ap     = 1+topt = y^{-theta} = exp(theta*e)
//      Z+1    = ap1 + ap2 - 1          (Newton residual is exactly 0)
//      logcur = -(1/theta+1)*log(Z+1) + (theta+1)*e
//    -> 6 hardware transcendentals/row (v_exp_f32 / v_log_f32), no powf.
//  * Memory-bound target: ~136 MB @ 23.3 TB/s ~= 5.8 us.
//  * Deterministic tree reductions (no float atomics).
// ---------------------------------------------------------------------------

typedef __attribute__((ext_vector_type(16))) _Float16 v16h;
typedef __attribute__((ext_vector_type(8)))  float    v8f;
typedef __attribute__((ext_vector_type(4)))  float    f32x4;

#define NBLOCKS        1024
#define NTHREADS       256
#define WAVES_PER_BLK  (NTHREADS / 32)

// d_ws layout:
//   bytes [0,256):   float wfold[64]  (wt[32], wc[32])
//   bytes [256,264): float bias[2]    (wfold[64], wfold[65])
//   bytes [512,...): double partials[NBLOCKS]

__global__ void fold_weights(const float* __restrict__ W1t, const float* __restrict__ b1t,
                             const float* __restrict__ W2t, const float* __restrict__ b2t,
                             const float* __restrict__ W1c, const float* __restrict__ b1c,
                             const float* __restrict__ W2c, const float* __restrict__ b2c,
                             float* __restrict__ wfold) {
    int tid = threadIdx.x;
    if (tid < 32) {
        float s = 0.f;
        #pragma unroll
        for (int h = 0; h < 32; ++h) s += W1t[tid * 32 + h] * W2t[h];
        wfold[tid] = s;
        if (tid == 0) {
            float b = b2t[0];
            for (int h = 0; h < 32; ++h) b += b1t[h] * W2t[h];
            wfold[64] = b;
        }
    } else {
        int f = tid - 32;
        float s = 0.f;
        #pragma unroll
        for (int h = 0; h < 32; ++h) s += W1c[f * 32 + h] * W2c[h];
        wfold[32 + f] = s;
        if (f == 0) {
            float b = b2c[0];
            for (int h = 0; h < 32; ++h) b += b1c[h] * W2c[h];
            wfold[65] = b;
        }
    }
}

__global__ __launch_bounds__(NTHREADS) void copula_main(
    const float* __restrict__ x, const float* __restrict__ tv, const float* __restrict__ cv,
    const float* __restrict__ wfold,
    const float* __restrict__ p_shape_t, const float* __restrict__ p_scale_t,
    const float* __restrict__ p_shape_c, const float* __restrict__ p_scale_c,
    const float* __restrict__ p_theta,
    double* __restrict__ partials, int tiles)
{
    __shared__ float  xchg[WAVES_PER_BLK * 64];
    __shared__ double wsum[WAVES_PER_BLK];

    const int tid    = threadIdx.x;
    const int lane   = tid & 31;
    const int wib    = tid >> 5;
    const int gwave  = (blockIdx.x * NTHREADS + tid) >> 5;
    const int nwaves = (gridDim.x * NTHREADS) >> 5;

    const float shape_t = *p_shape_t, scale_t = *p_scale_t;
    const float shape_c = *p_shape_c, scale_c = *p_scale_c;
    const float theta   = *p_theta;
    const float ls_t  = __logf(scale_t), ls_c  = __logf(scale_c);
    const float lsh_t = __logf(shape_t), lsh_c = __logf(shape_c);
    const float mexp  = -(1.0f / theta + 1.0f);   // exponent on (Z+1)
    const float thp1  = theta + 1.0f;             // factor on e = -logS

    const float bias_t = wfold[64];
    const float bias_c = wfold[65];

    // --- B matrix (32x16 f16): col 0 = folded t-weights, col 1 = folded
    // c-weights, cols 2..15 = 0. 16-bit B layout: lanes 0-15 = N, K = half*8+e (e<8), 16+half*8+e-8 (e>=8).
    const int colN = lane & 15;
    const int half = lane >> 4;      // 0: K 0-7,16-23; 1: K 8-15,24-31
    const int kb   = half * 16;
    v16h bmat;
    {
        const float* wv = wfold + (colN == 1 ? 32 : 0);
        const bool active = (colN < 2);
        #pragma unroll
        for (int e = 0; e < 16; ++e)
            bmat[e] = active ? (_Float16)wv[e < 8 ? half * 8 + e : 16 + half * 8 + (e - 8)] : (_Float16)0.0f;
    }

    // A layout (16-bit 16x32): lane = M%16 with lane-half selecting K groups:
    // elements 0..7 -> K = half*8 + e ; elements 8..15 -> K = 16 + half*8 + e.
    const int rlane = lane & 15;
    const int k1    = half * 8;

    float* wbuf = xchg + wib * 64;
    double acc = 0.0;

    for (int tI = gwave; tI < tiles; tI += nwaves) {
        const int r0 = tI * 32;

        // Prefetch next tile (global_prefetch_b8)
        const int tn = tI + nwaves;
        if (tn < tiles) {
            __builtin_prefetch(x + (size_t)(tn * 32 + rlane) * 32 + k1, 0, 0);
            __builtin_prefetch(x + (size_t)(tn * 32 + 16 + rlane) * 32 + k1, 0, 0);
        }

        // ---- load A tiles (rows r0..r0+15 and r0+16..r0+31), fp32 -> fp16
        v16h a0, a1;
        {
            const f32x4* p = (const f32x4*)(x + (size_t)(r0 + rlane) * 32 + k1);
            f32x4 u0 = p[0], u1 = p[1], u2 = p[4], u3 = p[5];
            const f32x4* q = (const f32x4*)(x + (size_t)(r0 + 16 + rlane) * 32 + k1);
            f32x4 w0 = q[0], w1 = q[1], w2 = q[4], w3 = q[5];
            #pragma unroll
            for (int e = 0; e < 4; ++e) {
                a0[e]      = (_Float16)u0[e];
                a0[4 + e]  = (_Float16)u1[e];
                a0[8 + e]  = (_Float16)u2[e];
                a0[12 + e] = (_Float16)u3[e];
                a1[e]      = (_Float16)w0[e];
                a1[4 + e]  = (_Float16)w1[e];
                a1[8 + e]  = (_Float16)w2[e];
                a1[12 + e] = (_Float16)w3[e];
            }
        }

        v8f cz = {0.f, 0.f, 0.f, 0.f, 0.f, 0.f, 0.f, 0.f};
        v8f d0 = __builtin_amdgcn_wmma_f32_16x16x32_f16(false, a0, false, bmat,
                                                        (short)0, cz, false, false);
        v8f d1 = __builtin_amdgcn_wmma_f32_16x16x32_f16(false, a1, false, bmat,
                                                        (short)0, cz, false, false);

        // ---- scatter betas: D layout lane(<16)=N, VGPR v -> M=v (+8 for hi half)
        // wbuf[col*32 + row] ; only N columns 0 (beta_t) and 1 (beta_c) matter.
        if (colN < 2) {
            float* pd = wbuf + colN * 32 + half * 8;
            #pragma unroll
            for (int v = 0; v < 8; ++v) pd[v] = d0[v];
            #pragma unroll
            for (int v = 0; v < 8; ++v) pd[16 + v] = d1[v];
        }
        asm volatile("s_wait_dscnt 0" ::: "memory");
        __builtin_amdgcn_wave_barrier();

        const float beta_t = wbuf[lane]      + bias_t;
        const float beta_c = wbuf[32 + lane] + bias_c;

        // ---- per-row survival-copula math (closed form, 6 hw transcendentals)
        const int row = r0 + lane;
        const float ti = tv[row];
        const float ci = cv[row];
        const float lt = __logf(ti);

        const float k_t = __builtin_fmaf(shape_t, lt - ls_t, beta_t);
        const float k_c = __builtin_fmaf(shape_c, lt - ls_c, beta_c);
        const float eE  = __expf(k_t);                 // = -logS_E
        const float eC  = __expf(k_c);                 // = -logS_C
        const float ld_t = k_t - lt + lsh_t - eE;      // log_density(event)
        const float ld_c = k_c - lt + lsh_c - eC;      // log_density(censor)

        // ap = 1+topt = y^{-theta} = exp(theta*e); Z+1 = ap1+ap2-1
        const float ap1 = __expf(theta * eE);
        const float ap2 = __expf(theta * eC);
        const float Zp1 = ap1 + ap2 - 1.0f;
        const float lZ  = __logf(Zp1);

        // log cur_j = -(1/theta+1)*log(Z+1) + (theta+1)*e_j
        const float logcur1 = __builtin_fmaf(mexp, lZ, thp1 * eE);
        const float logcur2 = __builtin_fmaf(mexp, lZ, thp1 * eC);

        const float logL = ci * (ld_t + logcur1) + (1.0f - ci) * (ld_c + logcur2);
        acc += (double)logL;
    }

    // ---- deterministic reduction: wave shuffle tree -> LDS -> block partial
    #pragma unroll
    for (int off = 16; off > 0; off >>= 1)
        acc += __shfl_xor(acc, off, 32);
    if (lane == 0) wsum[wib] = acc;
    __syncthreads();
    if (tid == 0) {
        double s = 0.0;
        #pragma unroll
        for (int i = 0; i < WAVES_PER_BLK; ++i) s += wsum[i];
        partials[blockIdx.x] = s;
    }
}

__global__ void final_reduce(const double* __restrict__ partials, int n,
                             float* __restrict__ out) {
    __shared__ double sh[256];
    double s = 0.0;
    for (int i = threadIdx.x; i < n; i += 256) s += partials[i];
    sh[threadIdx.x] = s;
    __syncthreads();
    for (int stride = 128; stride > 0; stride >>= 1) {
        if ((int)threadIdx.x < stride) sh[threadIdx.x] += sh[threadIdx.x + stride];
        __syncthreads();
    }
    if (threadIdx.x == 0) out[0] = (float)sh[0];
}

extern "C" void kernel_launch(void* const* d_in, const int* in_sizes, int n_in,
                              void* d_out, int out_size, void* d_ws, size_t ws_size,
                              hipStream_t stream) {
    const float* x       = (const float*)d_in[0];
    const float* t       = (const float*)d_in[1];
    const float* c       = (const float*)d_in[2];
    const float* W1t     = (const float*)d_in[3];
    const float* b1t     = (const float*)d_in[4];
    const float* W2t     = (const float*)d_in[5];
    const float* b2t     = (const float*)d_in[6];
    const float* W1c     = (const float*)d_in[7];
    const float* b1c     = (const float*)d_in[8];
    const float* W2c     = (const float*)d_in[9];
    const float* b2c     = (const float*)d_in[10];
    const float* shape_t = (const float*)d_in[11];
    const float* scale_t = (const float*)d_in[12];
    const float* shape_c = (const float*)d_in[13];
    const float* scale_c = (const float*)d_in[14];
    const float* theta   = (const float*)d_in[15];

    float*  wfold    = (float*)d_ws;
    double* partials = (double*)((char*)d_ws + 512);

    const int B = in_sizes[1];       // 1,000,000 (divisible by 32)
    const int tiles = B / 32;

    fold_weights<<<1, 64, 0, stream>>>(W1t, b1t, W2t, b2t, W1c, b1c, W2c, b2c, wfold);
    copula_main<<<NBLOCKS, NTHREADS, 0, stream>>>(x, t, c, wfold,
                                                  shape_t, scale_t, shape_c, scale_c,
                                                  theta, partials, tiles);
    final_reduce<<<1, 256, 0, stream>>>(partials, NBLOCKS, (float*)d_out);
}
